// LinkPredictor_77481210020193
// MI455X (gfx1250) — compile-verified, matrix operand order checked by reading, not run on hardware
//
#include <hip/hip_runtime.h>
#include <math.h>

typedef __attribute__((ext_vector_type(16))) _Float16 v16h;
typedef __attribute__((ext_vector_type(8)))  float    v8f;

#define N_NODES 50000
#define N_EDGES 1600000

// ---------------------------------------------------------------------------
// f32 -> f16 conversion kernels
// ---------------------------------------------------------------------------
__global__ void k_concat_f16(const float* __restrict__ xi,
                             const float* __restrict__ xj,
                             _Float16* __restrict__ out) {
  long long tid = (long long)blockIdx.x * blockDim.x + threadIdx.x;
  long long total = (long long)N_NODES * 1024;
  if (tid >= total) return;
  int n = (int)(tid >> 10);
  int c = (int)(tid & 1023);
  float v = (c < 512) ? xi[(long long)n * 512 + c]
                      : xj[(long long)n * 512 + (c - 512)];
  out[tid] = (_Float16)v;
}

__global__ void k_f32_to_f16(const float* __restrict__ in,
                             _Float16* __restrict__ out, int count) {
  int tid = blockIdx.x * blockDim.x + threadIdx.x;
  if (tid < count) out[tid] = (_Float16)in[tid];
}

// ---------------------------------------------------------------------------
// Register-blocked WMMA GEMM with double-buffered fragments.
// Each wave computes a 16 x (16*NB) strip: one A fragment per K-step feeds
// NB independent back-to-back WMMAs. Next K-step's fragments are loaded into
// the alternate buffer BEFORE the current WMMAs issue, so the scheduler can
// overlap global_load_b128 clauses with the XDL pipe (partial loadcnt waits).
//   Out = [relu](A[N,K] @ B[OUTC,K]^T [+ bias]),  f16 or f32 out.
// ---------------------------------------------------------------------------
template <int NB, bool RELU_BIAS, bool F16OUT>
__global__ void k_wmma_gemm(const _Float16* __restrict__ A,
                            const _Float16* __restrict__ B,
                            const float* __restrict__ bias,
                            void* __restrict__ Outv,
                            int K, int OUTC, int tilesNB, int totalWaves) {
  int wave = blockIdx.x * (blockDim.x >> 5) + (threadIdx.x >> 5);
  if (wave >= totalWaves) return;            // wave-uniform: EXEC stays all-1s
  int lane = threadIdx.x & 31;
  int m0 = (wave / tilesNB) << 4;
  int n0 = (wave % tilesNB) * (NB * 16);
  int rsel = lane & 15;
  int ksub = (lane >> 4) << 4;               // lanes 16-31 carry K+16..K+31
  const _Float16* ap = A + (size_t)(m0 + rsel) * K + ksub;
  const _Float16* bp[NB];
#pragma unroll
  for (int j = 0; j < NB; ++j)
    bp[j] = B + (size_t)(n0 + j * 16 + rsel) * K + ksub;

  v8f acc[NB];
#pragma unroll
  for (int j = 0; j < NB; ++j) acc[j] = (v8f){};

  // prologue: stage k=0 fragments
  v16h aC = *(const v16h*)ap;
  v16h bC[NB];
#pragma unroll
  for (int j = 0; j < NB; ++j) bC[j] = *(const v16h*)bp[j];

#pragma unroll 4
  for (int k = 32; k < K; k += 32) {
    // issue next-step loads first (separate buffer -> no wait-all)
    v16h aN = *(const v16h*)(ap + k);
    v16h bN[NB];
#pragma unroll
    for (int j = 0; j < NB; ++j) bN[j] = *(const v16h*)(bp[j] + k);
    // compute on previously loaded fragments
#pragma unroll
    for (int j = 0; j < NB; ++j)
      acc[j] = __builtin_amdgcn_wmma_f32_16x16x32_f16(false, aC, false, bC[j],
                                                      (short)0, acc[j], false, false);
    aC = aN;
#pragma unroll
    for (int j = 0; j < NB; ++j) bC[j] = bN[j];
  }
#pragma unroll
  for (int j = 0; j < NB; ++j)
    acc[j] = __builtin_amdgcn_wmma_f32_16x16x32_f16(false, aC, false, bC[j],
                                                    (short)0, acc[j], false, false);

  int mb = m0 + ((lane >> 4) << 3);
#pragma unroll
  for (int j = 0; j < NB; ++j) {
    int n = n0 + j * 16 + rsel;
    float bv = RELU_BIAS ? bias[n] : 0.0f;
#pragma unroll
    for (int r = 0; r < 8; ++r) {
      float v = acc[j][r];
      if (RELU_BIAS) v = fmaxf(v + bv, 0.0f);
      if (F16OUT)
        ((_Float16*)Outv)[(size_t)(mb + r) * OUTC + n] = (_Float16)v;
      else
        ((float*)Outv)[(size_t)(mb + r) * OUTC + n] = v;
    }
  }
}

// ---------------------------------------------------------------------------
// Fused residual block, register-blocked + double-buffered:
//   Out = relu(relu(A@W^T + b) + A@Aw^T), f16 out.
// Shared A fragment drives 2*NB WMMAs per K-step.
// ---------------------------------------------------------------------------
template <int NB>
__global__ void k_wmma_resblock(const _Float16* __restrict__ A,
                                const _Float16* __restrict__ Bw,
                                const _Float16* __restrict__ Ba,
                                const float* __restrict__ bias,
                                _Float16* __restrict__ Out,
                                int K, int OUTC, int tilesNB, int totalWaves) {
  int wave = blockIdx.x * (blockDim.x >> 5) + (threadIdx.x >> 5);
  if (wave >= totalWaves) return;
  int lane = threadIdx.x & 31;
  int m0 = (wave / tilesNB) << 4;
  int n0 = (wave % tilesNB) * (NB * 16);
  int rsel = lane & 15;
  int ksub = (lane >> 4) << 4;
  const _Float16* ap = A + (size_t)(m0 + rsel) * K + ksub;
  const _Float16* bwp[NB];
  const _Float16* bap[NB];
#pragma unroll
  for (int j = 0; j < NB; ++j) {
    bwp[j] = Bw + (size_t)(n0 + j * 16 + rsel) * K + ksub;
    bap[j] = Ba + (size_t)(n0 + j * 16 + rsel) * K + ksub;
  }

  v8f acc1[NB], acc2[NB];
#pragma unroll
  for (int j = 0; j < NB; ++j) { acc1[j] = (v8f){}; acc2[j] = (v8f){}; }

  v16h aC = *(const v16h*)ap;
  v16h b1C[NB], b2C[NB];
#pragma unroll
  for (int j = 0; j < NB; ++j) {
    b1C[j] = *(const v16h*)bwp[j];
    b2C[j] = *(const v16h*)bap[j];
  }

#pragma unroll 2
  for (int k = 32; k < K; k += 32) {
    v16h aN = *(const v16h*)(ap + k);
    v16h b1N[NB], b2N[NB];
#pragma unroll
    for (int j = 0; j < NB; ++j) {
      b1N[j] = *(const v16h*)(bwp[j] + k);
      b2N[j] = *(const v16h*)(bap[j] + k);
    }
#pragma unroll
    for (int j = 0; j < NB; ++j) {
      acc1[j] = __builtin_amdgcn_wmma_f32_16x16x32_f16(false, aC, false, b1C[j],
                                                       (short)0, acc1[j], false, false);
      acc2[j] = __builtin_amdgcn_wmma_f32_16x16x32_f16(false, aC, false, b2C[j],
                                                       (short)0, acc2[j], false, false);
    }
    aC = aN;
#pragma unroll
    for (int j = 0; j < NB; ++j) { b1C[j] = b1N[j]; b2C[j] = b2N[j]; }
  }
#pragma unroll
  for (int j = 0; j < NB; ++j) {
    acc1[j] = __builtin_amdgcn_wmma_f32_16x16x32_f16(false, aC, false, b1C[j],
                                                     (short)0, acc1[j], false, false);
    acc2[j] = __builtin_amdgcn_wmma_f32_16x16x32_f16(false, aC, false, b2C[j],
                                                     (short)0, acc2[j], false, false);
  }

  int mb = m0 + ((lane >> 4) << 3);
#pragma unroll
  for (int j = 0; j < NB; ++j) {
    int n = n0 + j * 16 + rsel;
    float bv = bias[n];
#pragma unroll
    for (int r = 0; r < 8; ++r) {
      float v = fmaxf(fmaxf(acc1[j][r] + bv, 0.0f) + acc2[j][r], 0.0f);
      Out[(size_t)(mb + r) * OUTC + n] = (_Float16)v;
    }
  }
}

// ---------------------------------------------------------------------------
// Graph aggregation (memory/atomic bound -> f32 scalar path)
// ---------------------------------------------------------------------------
__global__ void k_deg_init(float* __restrict__ deg) {
  int i = blockIdx.x * blockDim.x + threadIdx.x;
  if (i < N_NODES) deg[i] = 1.0f;              // +1 self loop
}

__global__ void k_deg_accum(const int* __restrict__ col, float* __restrict__ deg) {
  int e = blockIdx.x * blockDim.x + threadIdx.x;
  if (e < N_EDGES)
    __hip_atomic_fetch_add(&deg[col[e]], 1.0f, __ATOMIC_RELAXED,
                           __HIP_MEMORY_SCOPE_AGENT);
}

__global__ void k_dinv(const float* __restrict__ deg, float* __restrict__ dinv) {
  int i = blockIdx.x * blockDim.x + threadIdx.x;
  if (i < N_NODES) dinv[i] = rsqrtf(deg[i]);
}

// agg init = self-loop term: dinv[n]^2 * xw[n,:]
__global__ void k_selfloop(const float* __restrict__ xw,
                           const float* __restrict__ dinv,
                           float* __restrict__ agg) {
  long long t = (long long)blockIdx.x * blockDim.x + threadIdx.x;
  long long total = (long long)N_NODES * 128;
  if (t >= total) return;
  int n = (int)(t >> 7);
  float d = dinv[n];
  agg[t] = d * d * xw[t];
}

// one thread per (edge, 4 features): float4 gather + 4 hw float atomics
__global__ void k_edge_agg(const int* __restrict__ row, const int* __restrict__ col,
                           const float* __restrict__ dinv,
                           const float* __restrict__ xw,
                           float* __restrict__ agg) {
  long long t = (long long)blockIdx.x * blockDim.x + threadIdx.x;
  long long total = (long long)N_EDGES * 32;
  if (t >= total) return;
  int e = (int)(t >> 5);
  int f4 = (int)(t & 31) << 2;
  int r = row[e], c = col[e];
  float nrm = dinv[r] * dinv[c];
  const float4 v = *(const float4*)(xw + (size_t)r * 128 + f4);
  float* dst = agg + (size_t)c * 128 + f4;
  __hip_atomic_fetch_add(dst + 0, v.x * nrm, __ATOMIC_RELAXED, __HIP_MEMORY_SCOPE_AGENT);
  __hip_atomic_fetch_add(dst + 1, v.y * nrm, __ATOMIC_RELAXED, __HIP_MEMORY_SCOPE_AGENT);
  __hip_atomic_fetch_add(dst + 2, v.z * nrm, __ATOMIC_RELAXED, __HIP_MEMORY_SCOPE_AGENT);
  __hip_atomic_fetch_add(dst + 3, v.w * nrm, __ATOMIC_RELAXED, __HIP_MEMORY_SCOPE_AGENT);
}

// final: g = relu(agg + b_gcn); out = sigmoid(sigmoid(g . W_fc + b_fc))
// one wave per node: 4 features/lane, wave32 shuffle reduction
__global__ void k_final(const float* __restrict__ agg,
                        const float* __restrict__ bgcn,
                        const float* __restrict__ wfc,
                        const float* __restrict__ bfc,
                        float* __restrict__ out) {
  int wave = blockIdx.x * (blockDim.x >> 5) + (threadIdx.x >> 5);
  if (wave >= N_NODES) return;
  int lane = threadIdx.x & 31;
  int f = lane << 2;
  const float4 a = *(const float4*)(agg + (size_t)wave * 128 + f);
  const float4 b = *(const float4*)(bgcn + f);
  const float4 w = *(const float4*)(wfc + f);
  float s = fmaxf(a.x + b.x, 0.0f) * w.x + fmaxf(a.y + b.y, 0.0f) * w.y +
            fmaxf(a.z + b.z, 0.0f) * w.z + fmaxf(a.w + b.w, 0.0f) * w.w;
#pragma unroll
  for (int off = 16; off > 0; off >>= 1) s += __shfl_xor(s, off, 32);
  if (lane == 0) {
    float z = 1.0f / (1.0f + __expf(-(s + bfc[0])));
    out[wave] = 1.0f / (1.0f + __expf(-z));
  }
}

// ---------------------------------------------------------------------------
extern "C" void kernel_launch(void* const* d_in, const int* in_sizes, int n_in,
                              void* d_out, int out_size, void* d_ws, size_t ws_size,
                              hipStream_t stream) {
  const float* x_i   = (const float*)d_in[1];
  const float* x_j   = (const float*)d_in[2];
  const int*   eidx  = (const int*)d_in[3];
  const float* W_cat = (const float*)d_in[4];
  const float* b_cat = (const float*)d_in[5];
  const float* W1    = (const float*)d_in[6];
  const float* b1    = (const float*)d_in[7];
  const float* A1    = (const float*)d_in[8];
  const float* W2    = (const float*)d_in[9];
  const float* b2    = (const float*)d_in[10];
  const float* A2    = (const float*)d_in[11];
  const float* W_gcn = (const float*)d_in[12];
  const float* b_gcn = (const float*)d_in[13];
  const float* W_fc  = (const float*)d_in[14];
  const float* b_fc  = (const float*)d_in[15];
  const int* row = eidx;             // edge_index[0, :]
  const int* col = eidx + N_EDGES;   // edge_index[1, :]
  float* out = (float*)d_out;

  // workspace bump allocator (256B aligned blocks)
  char* wp = (char*)d_ws;
  auto alloc = [&](size_t bytes) -> void* {
    void* p = (void*)wp;
    wp += (bytes + 255) & ~(size_t)255;
    return p;
  };
  _Float16* Xcat = (_Float16*)alloc((size_t)N_NODES * 1024 * 2);
  _Float16* Wc_h = (_Float16*)alloc((size_t)512 * 1024 * 2);
  _Float16* H1   = (_Float16*)alloc((size_t)N_NODES * 512 * 2);
  _Float16* W1_h = (_Float16*)alloc((size_t)256 * 512 * 2);
  _Float16* A1_h = (_Float16*)alloc((size_t)256 * 512 * 2);
  _Float16* H2   = (_Float16*)alloc((size_t)N_NODES * 256 * 2);
  _Float16* W2_h = (_Float16*)alloc((size_t)128 * 256 * 2);
  _Float16* A2_h = (_Float16*)alloc((size_t)128 * 256 * 2);
  _Float16* H3   = (_Float16*)alloc((size_t)N_NODES * 128 * 2);
  _Float16* Wg_h = (_Float16*)alloc((size_t)128 * 128 * 2);
  float*    XW   = (float*)alloc((size_t)N_NODES * 128 * 4);
  float*    DEG  = (float*)alloc((size_t)N_NODES * 4);
  float*    DINV = (float*)alloc((size_t)N_NODES * 4);
  float*    AGG  = (float*)alloc((size_t)N_NODES * 128 * 4);

  const int B = 256;
  // --- conversions to f16 ---
  long long tcc = (long long)N_NODES * 1024;
  k_concat_f16<<<(int)((tcc + B - 1) / B), B, 0, stream>>>(x_i, x_j, Xcat);
  k_f32_to_f16<<<(512 * 1024 + B - 1) / B, B, 0, stream>>>(W_cat, Wc_h, 512 * 1024);
  k_f32_to_f16<<<(256 * 512 + B - 1) / B, B, 0, stream>>>(W1, W1_h, 256 * 512);
  k_f32_to_f16<<<(256 * 512 + B - 1) / B, B, 0, stream>>>(A1, A1_h, 256 * 512);
  k_f32_to_f16<<<(128 * 256 + B - 1) / B, B, 0, stream>>>(W2, W2_h, 128 * 256);
  k_f32_to_f16<<<(128 * 256 + B - 1) / B, B, 0, stream>>>(A2, A2_h, 128 * 256);
  k_f32_to_f16<<<(128 * 128 + B - 1) / B, B, 0, stream>>>(W_gcn, Wg_h, 128 * 128);

  const int WPB = 4;            // 4 wave32 per 128-thread block
  const int TM = N_NODES / 16;  // 3125 row tiles (exact)
  const int NB = 4;             // 16x64 strip per wave

  // --- layer 1: [N,1024] @ [512,1024]^T + b, relu ---
  {
    int tilesNB = (512 / 16) / NB, waves = TM * tilesNB;
    k_wmma_gemm<NB, true, true><<<(waves + WPB - 1) / WPB, WPB * 32, 0, stream>>>(
        Xcat, Wc_h, b_cat, H1, 1024, 512, tilesNB, waves);
  }
  // --- residual block 1: 512 -> 256 ---
  {
    int tilesNB = (256 / 16) / NB, waves = TM * tilesNB;
    k_wmma_resblock<NB><<<(waves + WPB - 1) / WPB, WPB * 32, 0, stream>>>(
        H1, W1_h, A1_h, b1, H2, 512, 256, tilesNB, waves);
  }
  // --- residual block 2: 256 -> 128 ---
  {
    int tilesNB = (128 / 16) / NB, waves = TM * tilesNB;
    k_wmma_resblock<NB><<<(waves + WPB - 1) / WPB, WPB * 32, 0, stream>>>(
        H2, W2_h, A2_h, b2, H3, 256, 128, tilesNB, waves);
  }
  // --- GCN matmul: XW = H3 @ W_gcn^T (f32 out, no bias/relu) ---
  {
    int tilesNB = (128 / 16) / NB, waves = TM * tilesNB;
    k_wmma_gemm<NB, false, false><<<(waves + WPB - 1) / WPB, WPB * 32, 0, stream>>>(
        H3, Wg_h, nullptr, XW, 128, 128, tilesNB, waves);
  }

  // --- degrees, normalization, aggregation ---
  k_deg_init<<<(N_NODES + B - 1) / B, B, 0, stream>>>(DEG);
  k_deg_accum<<<(N_EDGES + B - 1) / B, B, 0, stream>>>(col, DEG);
  k_dinv<<<(N_NODES + B - 1) / B, B, 0, stream>>>(DEG, DINV);
  long long tsl = (long long)N_NODES * 128;
  k_selfloop<<<(int)((tsl + B - 1) / B), B, 0, stream>>>(XW, DINV, AGG);
  long long tea = (long long)N_EDGES * 32;
  k_edge_agg<<<(int)((tea + B - 1) / B), B, 0, stream>>>(row, col, DINV, XW, AGG);

  // --- final: relu(agg + b_gcn), fc, double sigmoid ---
  k_final<<<(N_NODES * 32 + WPB * 32 - 1) / (WPB * 32), WPB * 32, 0, stream>>>(
      AGG, b_gcn, W_fc, b_fc, out);
}
